// OPTAttention_31860067402145
// MI455X (gfx1250) — matrix operand order repaired and checked by reading, not compiled, not run on hardware
//
#include <hip/hip_runtime.h>

// ---------------- problem constants ----------------
#define EDIM   2048
#define NHEADS 32
#define HDIM   64
#define BATCH  2
#define SEQ    2048
#define MTOT   (BATCH * SEQ)      // 4096 rows for all projections
#define SCALE_F 0.125f            // 64^-0.5

// ---------------- vector types ----------------
typedef __attribute__((ext_vector_type(16))) __bf16 bf16x16;
typedef __attribute__((ext_vector_type(8)))  __bf16 bf16x8;
typedef __attribute__((ext_vector_type(8)))  float  f32x8;
typedef __attribute__((ext_vector_type(4)))  float  f32x4;
typedef __attribute__((ext_vector_type(4)))  unsigned int uint32x4;
typedef __attribute__((ext_vector_type(8)))  unsigned int uint32x8;

__device__ __forceinline__ f32x8 wmma_bf16(bf16x16 a, bf16x16 b, f32x8 c) {
  // D = A(16x32 bf16) * B(32x16 bf16) + C(16x16 f32)
  return __builtin_amdgcn_wmma_f32_16x16x32_bf16(false, a, false, b, (short)0, c,
                                                 false, false);
}

// ---------------- fragment loaders (wave32 layouts per CDNA5 ISA 7.12.2) ----
// A 16x32 bf16, element (m,k) at base + m*ld + k.
__device__ __forceinline__ bf16x16 load_A_any(const __bf16* base, int ld) {
  int lane = threadIdx.x & 31;
  int m = lane & 15, hs = lane >> 4;
  const __bf16* row = base + (size_t)m * ld;
  union { bf16x16 f; struct { bf16x8 lo, hi; } p; } u;
  u.p.lo = *(const bf16x8*)(row + hs * 8);
  u.p.hi = *(const bf16x8*)(row + 16 + hs * 8);
  return u.f;
}

// B 32x16 bf16 where output column n is a contiguous run in memory:
// element (k,n) at base + n*ld + k  (covers W[N,K] used as W^T, and K^T in QK).
__device__ __forceinline__ bf16x16 load_B_kcontig(const __bf16* base, int ld) {
  int lane = threadIdx.x & 31;
  int n = lane & 15, hs = lane >> 4;
  const __bf16* row = base + (size_t)n * ld + hs * 8;
  union { bf16x16 f; struct { bf16x8 lo, hi; } p; } u;
  u.p.lo = *(const bf16x8*)(row);
  u.p.hi = *(const bf16x8*)(row + 16);
  return u.f;
}

// B 32x16 bf16 from a row-major LDS tile: element (k,n) at base + k*ld + n.
__device__ __forceinline__ bf16x16 load_B_lds_rowmajor(const __bf16* base, int ld) {
  int lane = threadIdx.x & 31;
  int n = lane & 15, hs = lane >> 4;
  bf16x16 f;
#pragma unroll
  for (int e = 0; e < 16; ++e) f[e] = base[((e & 8) * 2 + hs * 8 + (e & 7)) * ld + n];
  return f;
}

// A 16x32 sourced from f32 LDS probabilities, converted to bf16 in-register.
__device__ __forceinline__ bf16x16 load_A_ldsf32(const float* base, int ld) {
  int lane = threadIdx.x & 31;
  int m = lane & 15, hs = lane >> 4;
  const float* row = base + m * ld;
  bf16x16 f;
#pragma unroll
  for (int e = 0; e < 8; ++e) {
    f[e]     = (__bf16)row[hs * 8 + e];
    f[8 + e] = (__bf16)row[16 + hs * 8 + e];
  }
  return f;
}

// ---------------- async load global -> LDS (ASYNCcnt path) ------------------
// GVS mode: addr = SADDR(64b sgpr) + VADDR(i32 byte offset); VDST = LDS addr.
__device__ __forceinline__ void async_ld_b128(unsigned int lds_addr,
                                              unsigned int voff,
                                              const void* sbase) {
  asm volatile("global_load_async_to_lds_b128 %0, %1, %2"
               :: "v"(lds_addr), "v"(voff), "s"(sbase) : "memory");
}
__device__ __forceinline__ void wait_async0() {
  asm volatile("s_wait_asynccnt 0x0" ::: "memory");
}
__device__ __forceinline__ void wait_ds0() {
  asm volatile("s_wait_dscnt 0x0" ::: "memory");
}

// ---------------- TDM: 2-D bf16 tile (rows x 32) global -> LDS --------------
// Builds a D# (groups 0+1, 2-D tensor) and issues tensor_load_to_lds.
__device__ __forceinline__ void tdm_load_tile(unsigned int lds_off,
                                              const __bf16* gptr,
                                              unsigned int ld_elems,
                                              unsigned int rows) {
  unsigned long long ga = (unsigned long long)(uintptr_t)gptr;
  uint32x4 g0;
  g0[0] = 1u;                                          // count=1, no gather
  g0[1] = lds_off;                                     // lds_addr
  g0[2] = (unsigned int)ga;                            // global_addr[31:0]
  g0[3] = (unsigned int)((ga >> 32) & 0x01FFFFFFu)     // global_addr[56:32]
          | (2u << 30);                                // type=2 (image)
  uint32x8 g1;
  g1[0] = (1u << 16);                                  // data_size=1 -> 2 bytes
  g1[1] = (ld_elems & 0xFFFFu) << 16;                  // tensor_dim0[15:0]
  g1[2] = ((ld_elems >> 16) & 0xFFFFu);                // tensor_dim0[31:16]
  g1[3] = 16u                                          // tensor_dim1 = 2^20
          | (32u << 16);                               // tile_dim0 = 32
  g1[4] = (rows & 0xFFFFu);                            // tile_dim1 ; tile_dim2=0
  g1[5] = ld_elems;                                    // tensor_dim0_stride[31:0]
  g1[6] = 0u;
  g1[7] = 0u;
  asm volatile("tensor_load_to_lds %0, %1" :: "s"(g0), "s"(g1) : "memory");
}

// ---------------- cast kernel: f32 -> bf16, 8 elements/thread ----------------
__global__ void cast8_f32_bf16(const float* __restrict__ src,
                               __bf16* __restrict__ dst, int n8) {
  int i = blockIdx.x * blockDim.x + threadIdx.x;
  if (i >= n8) return;
  f32x4 a = *(const f32x4*)(src + (size_t)i * 8);
  f32x4 b = *(const f32x4*)(src + (size_t)i * 8 + 4);
  bf16x8 o;
#pragma unroll
  for (int e = 0; e < 4; ++e) { o[e] = (__bf16)a[e]; o[4 + e] = (__bf16)b[e]; }
  *(bf16x8*)(dst + (size_t)i * 8) = o;
}

// ---------------- WMMA GEMM with TDM double buffering -----------------------
// out[M,N] = A[M,K] @ W[N,K]^T + bias.  block = 8 waves, tile 128x128.
template <bool OUT_F32>
__global__ __launch_bounds__(256) void gemm_wmma_tdm(
    const __bf16* __restrict__ A, const __bf16* __restrict__ W,
    const float* __restrict__ bias, __bf16* __restrict__ outb,
    float* __restrict__ outf, int M, int N, int K) {
  __shared__ __bf16 sA[2][128 * 32];   // 8 KB per buffer
  __shared__ __bf16 sB[2][128 * 32];
  int lane = threadIdx.x & 31;
  int wave = threadIdx.x >> 5;
  int mblk = blockIdx.y * 128;
  int nblk = blockIdx.x * 128;
  const int KSTEPS = K / 32;
  bool issuer = (wave == 0);

  unsigned int aLds[2] = { (unsigned int)(uintptr_t)&sA[0][0],
                           (unsigned int)(uintptr_t)&sA[1][0] };
  unsigned int bLds[2] = { (unsigned int)(uintptr_t)&sB[0][0],
                           (unsigned int)(uintptr_t)&sB[1][0] };

  f32x8 acc[8] = {};

  if (issuer) {
    tdm_load_tile(aLds[0], A + (size_t)mblk * K, (unsigned)K, 128u);
    tdm_load_tile(bLds[0], W + (size_t)nblk * K, (unsigned)K, 128u);
  }
  for (int ks = 0; ks < KSTEPS; ++ks) {
    int cur = ks & 1;
    if (issuer) {
      if (ks + 1 < KSTEPS) {
        int kn = (ks + 1) * 32;
        tdm_load_tile(aLds[cur ^ 1], A + (size_t)mblk * K + kn, (unsigned)K, 128u);
        tdm_load_tile(bLds[cur ^ 1], W + (size_t)nblk * K + kn, (unsigned)K, 128u);
        __builtin_amdgcn_s_wait_tensorcnt((short)2);  // current tiles landed
      } else {
        __builtin_amdgcn_s_wait_tensorcnt((short)0);
      }
    }
    __syncthreads();                                  // tiles visible to all waves
    bf16x16 af = load_A_any(&sA[cur][0] + wave * 16 * 32, 32);
#pragma unroll
    for (int t = 0; t < 8; ++t) {
      bf16x16 bfrag = load_B_kcontig(&sB[cur][0] + t * 16 * 32, 32);
      acc[t] = wmma_bf16(af, bfrag, acc[t]);
    }
    __syncthreads();                                  // done reading before overwrite
  }

  int mbase = mblk + wave * 16;
  int n0 = lane & 15, hs = lane >> 4;
#pragma unroll
  for (int t = 0; t < 8; ++t) {
    int n = nblk + t * 16 + n0;
    float bv = bias[n];
#pragma unroll
    for (int r = 0; r < 8; ++r) {
      int m = mbase + r + hs * 8;
      float v = acc[t][r] + bv;
      if (OUT_F32) outf[(size_t)m * N + n] = v;
      else         outb[(size_t)m * N + n] = (__bf16)v;
    }
  }
}

// ---------------- flash attention: one wave per (b, h, 16-query tile) -------
// Per 128-key chunk: async-DMA the 128x64 V tile into LDS (overlapped with the
// score WMMAs and softmax), QK^T via WMMA, online softmax in LDS, P@V via WMMA
// with B-fragments gathered from the LDS V tile.
__global__ __launch_bounds__(32) void attn_wmma(const __bf16* __restrict__ Q,
                                                const __bf16* __restrict__ Kmat,
                                                const __bf16* __restrict__ V,
                                                __bf16* __restrict__ O) {
  __shared__ float  sc[16][128];     // score / prob chunk (8 KB)
  __shared__ __bf16 sV[128 * 64];    // V chunk, row-major (16 KB)
  __shared__ float rowmax[16], rowsum[16], alpha_s[16];

  int lane = threadIdx.x;
  int qt = blockIdx.x;              // query tile 0..127
  int hh = blockIdx.y;              // head
  int bb = blockIdx.z;              // batch
  int qbase = qt * 16;
  int n0 = lane & 15, hs = lane >> 4;

  const __bf16* Qb = Q + ((size_t)(bb * SEQ + qbase)) * EDIM + hh * HDIM;
  bf16x16 qa0 = load_A_any(Qb, EDIM);       // Q 16x64 as two 16x32 A-frags
  bf16x16 qa1 = load_A_any(Qb + 32, EDIM);

  const __bf16* Vhead = V + ((size_t)bb * SEQ) * EDIM + hh * HDIM;
  unsigned int sVbase = (unsigned int)(uintptr_t)&sV[0];

  f32x8 o[4] = {};                          // 16x64 output accumulator
  if (lane < 16) { rowmax[lane] = -3.0e38f; rowsum[lane] = 0.0f; }
  __syncthreads();

  for (int kc = 0; kc <= qbase; kc += 128) {
    int kcl = min(128, qbase + 16 - kc);    // valid keys this chunk
    int ntiles = (kcl + 15) >> 4;

    // ---- kick off async DMA of this chunk's V tile (overlaps QK + softmax) --
    wait_ds0();                              // prior chunk's LDS reads drained
#pragma unroll
    for (int p = 0; p < 32; ++p) {
      int idx = p * 32 + lane;               // 0..1023 16-byte pieces
      int r = idx >> 3;                      // row 0..127
      int c = idx & 7;                       // 16B piece within 128B row
      int row = kc + r; if (row >= SEQ) row = SEQ - 1;   // clamped, P==0 there
      unsigned int voff = (unsigned int)row * (EDIM * 2) + (unsigned int)(c * 16);
      async_ld_b128(sVbase + (unsigned int)(r * 128 + c * 16), voff, Vhead);
    }

    // ---- scores -> LDS (scaled, causal-masked) ----
    for (int kt = 0; kt < 8; ++kt) {
      if (kt < ntiles) {
        int kb = kc + kt * 16;
        const __bf16* Kb = Kmat + ((size_t)(bb * SEQ + kb)) * EDIM + hh * HDIM;
        f32x8 s = {};
        s = wmma_bf16(qa0, load_B_kcontig(Kb, EDIM), s);
        s = wmma_bf16(qa1, load_B_kcontig(Kb + 32, EDIM), s);
#pragma unroll
        for (int r = 0; r < 8; ++r) {
          int m = r + hs * 8;
          float v = s[r] * SCALE_F;
          if (kb + n0 > qbase + m) v = -3.0e38f;   // causal mask
          sc[m][kt * 16 + n0] = v;
        }
      } else {
#pragma unroll
        for (int r = 0; r < 8; ++r) sc[r + hs * 8][kt * 16 + n0] = -3.0e38f;
      }
    }
    __syncthreads();
    // ---- online softmax: chunk max -> running max, alpha ----
    for (int m = 0; m < 16; ++m) {
      float v = -3.0e38f;
      for (int j = lane; j < 128; j += 32) v = fmaxf(v, sc[m][j]);
      for (int off = 16; off; off >>= 1) v = fmaxf(v, __shfl_xor(v, off));
      if (lane == 0) {
        float mo = rowmax[m];
        float mn = fmaxf(mo, v);
        alpha_s[m] = __expf(mo - mn);
        rowmax[m] = mn;
      }
    }
    __syncthreads();
    // ---- exponentiate in LDS + running row sums ----
    for (int m = 0; m < 16; ++m) {
      float mx = rowmax[m];
      float ssum = 0.0f;
      for (int j = lane; j < 128; j += 32) {
        float e = __expf(sc[m][j] - mx);
        sc[m][j] = e;
        ssum += e;
      }
      for (int off = 16; off; off >>= 1) ssum += __shfl_xor(ssum, off);
      if (lane == 0) rowsum[m] = rowsum[m] * alpha_s[m] + ssum;
    }
    __syncthreads();
    // ---- rescale running output by alpha ----
#pragma unroll
    for (int t = 0; t < 4; ++t)
#pragma unroll
      for (int r = 0; r < 8; ++r) o[t][r] *= alpha_s[r + hs * 8];

    // ---- V tile must be in LDS now ----
    wait_async0();
    __syncthreads();

    // ---- P @ V via WMMA, B-fragments gathered from LDS V tile ----
    int nk = (kcl + 31) >> 5;
    for (int ks = 0; ks < nk; ++ks) {
      bf16x16 pa = load_A_ldsf32(&sc[0][ks * 32], 128);
#pragma unroll
      for (int t = 0; t < 4; ++t)
        o[t] = wmma_bf16(pa, load_B_lds_rowmajor(&sV[(ks * 32) * 64 + t * 16], 64),
                         o[t]);
    }
    __syncthreads();
  }

  // ---- normalize and store 16x64 bf16 tile ----
  __bf16* Ob = O + ((size_t)(bb * SEQ + qbase)) * EDIM + hh * HDIM;
#pragma unroll
  for (int t = 0; t < 4; ++t)
#pragma unroll
    for (int r = 0; r < 8; ++r) {
      int m = r + hs * 8;
      float v = o[t][r] / rowsum[m];
      Ob[(size_t)m * EDIM + t * 16 + n0] = (__bf16)v;
    }
}

// ---------------- host launcher ----------------
extern "C" void kernel_launch(void* const* d_in, const int* in_sizes, int n_in,
                              void* d_out, int out_size, void* d_ws, size_t ws_size,
                              hipStream_t stream) {
  (void)in_sizes; (void)n_in; (void)out_size; (void)ws_size;
  const float* h  = (const float*)d_in[0];
  const float* Wq = (const float*)d_in[1];
  const float* bq = (const float*)d_in[2];
  const float* Wk = (const float*)d_in[3];
  const float* bk = (const float*)d_in[4];
  const float* Wv = (const float*)d_in[5];
  const float* bv = (const float*)d_in[6];
  const float* Wo = (const float*)d_in[7];
  const float* bo = (const float*)d_in[8];
  float* out = (float*)d_out;

  char* ws = (char*)d_ws;
  size_t off = 0;
  __bf16* hbf = (__bf16*)(ws + off); off += (size_t)MTOT * EDIM * 2;
  __bf16* wqb = (__bf16*)(ws + off); off += (size_t)EDIM * EDIM * 2;
  __bf16* wkb = (__bf16*)(ws + off); off += (size_t)EDIM * EDIM * 2;
  __bf16* wvb = (__bf16*)(ws + off); off += (size_t)EDIM * EDIM * 2;
  __bf16* wob = (__bf16*)(ws + off); off += (size_t)EDIM * EDIM * 2;
  __bf16* Qbf = (__bf16*)(ws + off); off += (size_t)MTOT * EDIM * 2;
  __bf16* Kbf = (__bf16*)(ws + off); off += (size_t)MTOT * EDIM * 2;
  __bf16* Vbf = (__bf16*)(ws + off); off += (size_t)MTOT * EDIM * 2;
  __bf16* Obf = (__bf16*)(ws + off); off += (size_t)MTOT * EDIM * 2;

  // 1) cast inputs to bf16
  {
    int thr = 256;
    int n8h = (MTOT * EDIM) / 8;
    int n8w = (EDIM * EDIM) / 8;
    cast8_f32_bf16<<<(n8h + thr - 1) / thr, thr, 0, stream>>>(h,  hbf, n8h);
    cast8_f32_bf16<<<(n8w + thr - 1) / thr, thr, 0, stream>>>(Wq, wqb, n8w);
    cast8_f32_bf16<<<(n8w + thr - 1) / thr, thr, 0, stream>>>(Wk, wkb, n8w);
    cast8_f32_bf16<<<(n8w + thr - 1) / thr, thr, 0, stream>>>(Wv, wvb, n8w);
    cast8_f32_bf16<<<(n8w + thr - 1) / thr, thr, 0, stream>>>(Wo, wob, n8w);
  }

  // 2) QKV projections (bf16 out, f32 accumulate + bias), TDM-fed
  dim3 gblk(256);
  dim3 ggrid(EDIM / 128, MTOT / 128);   // (16, 32)
  gemm_wmma_tdm<false><<<ggrid, gblk, 0, stream>>>(hbf, wqb, bq, Qbf, nullptr,
                                                   MTOT, EDIM, EDIM);
  gemm_wmma_tdm<false><<<ggrid, gblk, 0, stream>>>(hbf, wkb, bk, Kbf, nullptr,
                                                   MTOT, EDIM, EDIM);
  gemm_wmma_tdm<false><<<ggrid, gblk, 0, stream>>>(hbf, wvb, bv, Vbf, nullptr,
                                                   MTOT, EDIM, EDIM);

  // 3) causal flash attention (async V staging + WMMA)
  dim3 agrid(SEQ / 16, NHEADS, BATCH); // (128, 32, 2)
  attn_wmma<<<agrid, 32, 0, stream>>>(Qbf, Kbf, Vbf, Obf);

  // 4) output projection -> f32 d_out
  gemm_wmma_tdm<true><<<ggrid, gblk, 0, stream>>>(Obf, wob, bo, nullptr, out,
                                                  MTOT, EDIM, EDIM);
}
